// MultiHeadAttention_79534204387618
// MI455X (gfx1250) — matrix operand order repaired and checked by reading, not compiled, not run on hardware
//
#include <hip/hip_runtime.h>
#include <hip/hip_bf16.h>

// MHA: D=1024, H=16, HD=64, LQ=LK=2048.  All matmuls via v_wmma_f32_16x16x32_bf16.
// Pipeline: fp32->bf16 convert; Q=Xe@Wq^T*0.125; K=Hh@Wk^T; Vt=transpose(Hh@Wv^T);
//           flash-attention per (head, 16-query block); out(f32) = ctx@Wo^T.
// Workspace: 16M bf16 elements = 32 MB.

#define D_  1024
#define H_  16
#define HD_ 64
#define LQ_ 2048
#define LK_ 2048

typedef __bf16 bf16_t;
typedef __attribute__((ext_vector_type(16))) __bf16 v16bf;
typedef __attribute__((ext_vector_type(8)))  __bf16 v8bf;
typedef __attribute__((ext_vector_type(4)))  __bf16 v4bf;
typedef __attribute__((ext_vector_type(8)))  float  v8f;
typedef __attribute__((ext_vector_type(4)))  float  v4f;

__device__ __forceinline__ v8f zero8() {
  v8f z = {0.f, 0.f, 0.f, 0.f, 0.f, 0.f, 0.f, 0.f};
  return z;
}

// A-fragment (16x32 bf16, row selected by lane&15 outside):
// lanes 0-15 hold K {0..7, 16..23}; lanes 16-31 hold K {8..15, 24..31}.
__device__ __forceinline__ v16bf load_a_frag(const bf16_t* p, int lane) {
  const int off = (lane & 16) ? 8 : 0;
  v8bf lo = *(const v8bf*)(p + off);
  v8bf hi = *(const v8bf*)(p + off + 16);
  v16bf r;
#pragma unroll
  for (int i = 0; i < 8; ++i) { r[i] = lo[i]; r[i + 8] = hi[i]; }
  return r;
}

// B-fragment (32x16 bf16, column selected by lane&15 outside):
// lanes 0-15 hold K {0..7, 16..23}; lanes 16-31 hold K {8..15, 24..31}.
__device__ __forceinline__ v16bf load_b_frag(const bf16_t* p, int lane) {
  const int off = (lane & 16) ? 8 : 0;
  v8bf lo = *(const v8bf*)(p + off);
  v8bf hi = *(const v8bf*)(p + off + 16);
  v16bf r;
#pragma unroll
  for (int i = 0; i < 8; ++i) { r[i] = lo[i]; r[i + 8] = hi[i]; }
  return r;
}

__global__ void f32_to_bf16_kernel(const float* __restrict__ in,
                                   bf16_t* __restrict__ out, int n4) {
  int i = blockIdx.x * blockDim.x + threadIdx.x;
  int stride = gridDim.x * blockDim.x;
  const v4f* in4 = (const v4f*)in;
  v4bf* out4 = (v4bf*)out;
  for (; i < n4; i += stride) {
    v4f x = in4[i];
    v4bf y;
#pragma unroll
    for (int j = 0; j < 4; ++j) y[j] = (bf16_t)x[j];
    out4[i] = y;
  }
}

struct Frags {
  v16bf a0, a1;
  v16bf b[4];
};

__device__ __forceinline__ void load_frags(Frags& f, const bf16_t* arow0,
                                           const bf16_t* arow1,
                                           const bf16_t* const* wrow, int kk,
                                           int lane) {
  f.a0 = load_a_frag(arow0 + kk, lane);
  f.a1 = load_a_frag(arow1 + kk, lane);
#pragma unroll
  for (int c = 0; c < 4; ++c) f.b[c] = load_b_frag(wrow[c] + kk, lane);
}

__device__ __forceinline__ void mma_frags(v8f acc[8], const Frags& f) {
#pragma unroll
  for (int c = 0; c < 4; ++c) {
    acc[c] = __builtin_amdgcn_wmma_f32_16x16x32_bf16(
        false, f.a0, false, f.b[c], (short)0, acc[c], false, false);
    acc[c + 4] = __builtin_amdgcn_wmma_f32_16x16x32_bf16(
        false, f.a1, false, f.b[c], (short)0, acc[c + 4], false, false);
  }
}

// C = A[MxK] @ W[NxK]^T, bf16 in, f32 accumulate.  One wave per 32x64 tile,
// k-loop unrolled x2 with two explicit fragment buffers (no rotation copies).
// Requires K/32 even.
// mode 0: store bf16 row-major [M][N]; mode 1: store bf16 transposed [N][M] (Vt);
// mode 2: store f32 row-major [M][N].
__global__ __launch_bounds__(256) void gemm_bf16_wmma(
    const bf16_t* __restrict__ A, const bf16_t* __restrict__ W,
    bf16_t* __restrict__ outb, float* __restrict__ outf,
    int M, int N, int K, float scale, int mode) {
  const int lane = threadIdx.x & 31;
  const int wave = threadIdx.x >> 5;
  const int gw = blockIdx.x * 8 + wave;
  const int tilesN = N >> 6;
  const int mt = gw / tilesN;  // 32-row tile index
  const int nt = gw % tilesN;
  if (mt * 32 >= M) return;  // wave-uniform: EXEC stays all-ones for active waves
  const int l15 = lane & 15;

  const bf16_t* arow0 = A + (size_t)(mt * 32 + l15) * K;
  const bf16_t* arow1 = arow0 + (size_t)16 * K;
  const bf16_t* wrow[4];
#pragma unroll
  for (int c = 0; c < 4; ++c)
    wrow[c] = W + (size_t)(nt * 64 + c * 16 + l15) * K;

  v8f acc[8];
#pragma unroll
  for (int c = 0; c < 8; ++c) acc[c] = zero8();

  Frags f0, f1;
  load_frags(f0, arow0, arow1, wrow, 0, lane);

  // Each iteration retires two k-steps (f0 then f1) while prefetching the
  // following two; buffers alternate so no register-rotation copies occur.
  int kk = 32;
  for (; kk + 32 < K; kk += 64) {
    load_frags(f1, arow0, arow1, wrow, kk, lane);
    mma_frags(acc, f0);
    load_frags(f0, arow0, arow1, wrow, kk + 32, lane);
    mma_frags(acc, f1);
  }
  // K/32 even: exactly one k-step left (in-flight buffer f0 holds K-64).
  load_frags(f1, arow0, arow1, wrow, K - 32, lane);
  mma_frags(acc, f0);
  mma_frags(acc, f1);

#pragma unroll
  for (int half = 0; half < 2; ++half) {
    const int r0 = mt * 32 + half * 16 + ((lane & 16) ? 8 : 0);
#pragma unroll
    for (int c = 0; c < 4; ++c) {
      const int cc = nt * 64 + c * 16 + l15;
#pragma unroll
      for (int r = 0; r < 8; ++r) {
        const int rr = r0 + r;
        const float v = acc[half * 4 + c][r] * scale;
        if (mode == 2)      outf[(size_t)rr * N + cc] = v;
        else if (mode == 1) outb[(size_t)cc * M + rr] = (bf16_t)v;
        else                outb[(size_t)rr * N + cc] = (bf16_t)v;
      }
    }
  }
}

// Flash attention: one wave per (head, 16-query block).  Q pre-scaled by 1/8.
// Kb row-major [LK][D]; Vt transposed [(h*64+d)][LK]; Ctx bf16 [LQ][D].
__global__ __launch_bounds__(256) void flash_attn_wmma(
    const bf16_t* __restrict__ Q, const bf16_t* __restrict__ Kb,
    const bf16_t* __restrict__ Vt, bf16_t* __restrict__ Ctx) {
  __shared__ bf16_t pbuf[8][16 * 32];  // per-wave P scratch (C-layout -> A-layout)
  const int lane = threadIdx.x & 31;
  const int wave = threadIdx.x >> 5;
  const int gw = blockIdx.x * 8 + wave;
  const int head = gw >> 7;   // LQ/16 = 128 query blocks per head
  const int qb = gw & 127;
  const int l15 = lane & 15;
  bf16_t* pw = &pbuf[wave][0];

  // Q fragments for K-dim d=0..63 (two K=32 chunks), held for the whole loop.
  const bf16_t* qrow = Q + (size_t)(qb * 16 + l15) * D_ + head * HD_;
  v16bf qa0 = load_a_frag(qrow, lane);
  v16bf qa1 = load_a_frag(qrow + 32, lane);

  v8f acc[4];
#pragma unroll
  for (int c = 0; c < 4; ++c) acc[c] = zero8();
  float m[8], l[8];
#pragma unroll
  for (int r = 0; r < 8; ++r) { m[r] = -3.0e38f; l[r] = 0.f; }

  for (int kb = 0; kb < LK_; kb += 32) {
    // S = Q @ K^T  (two 16x16 key tiles, reduce over d in two K=32 WMMAs)
    v8f s[2];
#pragma unroll
    for (int t = 0; t < 2; ++t) {
      const bf16_t* krow = Kb + (size_t)(kb + t * 16 + l15) * D_ + head * HD_;
      v16bf b0 = load_b_frag(krow, lane);
      v16bf b1 = load_b_frag(krow + 32, lane);
      v8f z = zero8();
      v8f sp = __builtin_amdgcn_wmma_f32_16x16x32_bf16(
          false, qa0, false, b0, (short)0, z, false, false);
      s[t] = __builtin_amdgcn_wmma_f32_16x16x32_bf16(
          false, qa1, false, b1, (short)0, sp, false, false);
    }

    // Issue V-fragment loads now: independent of softmax, so the exp/shuffle
    // VALU work below runs under the load shadow.
    v16bf bv[4];
#pragma unroll
    for (int c = 0; c < 4; ++c) {
      const bf16_t* vcol = Vt + (size_t)(head * HD_ + c * 16 + l15) * LK_ + kb;
      bv[c] = load_b_frag(vcol, lane);
    }

    // Online softmax: row = (C-layout) VGPR index r within each lane-half.
    float mn[8], alpha[8];
#pragma unroll
    for (int r = 0; r < 8; ++r) {
      float v = fmaxf(s[0][r], s[1][r]);
      v = fmaxf(v, __shfl_xor(v, 1, 32));
      v = fmaxf(v, __shfl_xor(v, 2, 32));
      v = fmaxf(v, __shfl_xor(v, 4, 32));
      v = fmaxf(v, __shfl_xor(v, 8, 32));
      mn[r] = fmaxf(m[r], v);
      alpha[r] = __expf(m[r] - mn[r]);
      m[r] = mn[r];
    }
    float p0[8], p1[8];
#pragma unroll
    for (int r = 0; r < 8; ++r) {
      p0[r] = __expf(s[0][r] - mn[r]);
      p1[r] = __expf(s[1][r] - mn[r]);
      float rs = p0[r] + p1[r];
      rs += __shfl_xor(rs, 1, 32);
      rs += __shfl_xor(rs, 2, 32);
      rs += __shfl_xor(rs, 4, 32);
      rs += __shfl_xor(rs, 8, 32);
      l[r] = l[r] * alpha[r] + rs;
    }
#pragma unroll
    for (int c = 0; c < 4; ++c)
#pragma unroll
      for (int r = 0; r < 8; ++r) acc[c][r] *= alpha[r];

    // P (C-layout f32) -> LDS row-major bf16 [16][32]; re-read as A-fragment.
    const int prow = (lane & 16) ? 8 : 0;
#pragma unroll
    for (int r = 0; r < 8; ++r) {
      pw[(prow + r) * 32 + l15]      = (bf16_t)p0[r];
      pw[(prow + r) * 32 + 16 + l15] = (bf16_t)p1[r];
    }
    v16bf pa = load_a_frag(pw + l15 * 32, lane);  // DS in-order within wave

    // acc += P @ V
#pragma unroll
    for (int c = 0; c < 4; ++c) {
      acc[c] = __builtin_amdgcn_wmma_f32_16x16x32_bf16(
          false, pa, false, bv[c], (short)0, acc[c], false, false);
    }
  }

  // ctx = acc / l, store bf16 for the output projection GEMM.
  float inv[8];
#pragma unroll
  for (int r = 0; r < 8; ++r) inv[r] = 1.0f / l[r];
  const int r0 = qb * 16 + ((lane & 16) ? 8 : 0);
#pragma unroll
  for (int c = 0; c < 4; ++c) {
    const int col = head * HD_ + c * 16 + l15;
#pragma unroll
    for (int r = 0; r < 8; ++r)
      Ctx[(size_t)(r0 + r) * D_ + col] = (bf16_t)(acc[c][r] * inv[r]);
  }
}

extern "C" void kernel_launch(void* const* d_in, const int* in_sizes, int n_in,
                              void* d_out, int out_size, void* d_ws, size_t ws_size,
                              hipStream_t stream) {
  const float* Xe = (const float*)d_in[0];
  const float* Hh = (const float*)d_in[1];
  const float* Wq = (const float*)d_in[2];
  const float* Wk = (const float*)d_in[3];
  const float* Wv = (const float*)d_in[4];
  const float* Wo = (const float*)d_in[5];
  float* out = (float*)d_out;

  bf16_t* p = (bf16_t*)d_ws;
  bf16_t* Xeb = p; p += (size_t)LQ_ * D_;
  bf16_t* Hhb = p; p += (size_t)LK_ * D_;
  bf16_t* Wqb = p; p += (size_t)D_ * D_;
  bf16_t* Wkb = p; p += (size_t)D_ * D_;
  bf16_t* Wvb = p; p += (size_t)D_ * D_;
  bf16_t* Wob = p; p += (size_t)D_ * D_;
  bf16_t* Qb  = p; p += (size_t)LQ_ * D_;
  bf16_t* Kbb = p; p += (size_t)LK_ * D_;
  bf16_t* Vtb = p; p += (size_t)LK_ * D_;
  bf16_t* Ctx = p; p += (size_t)LQ_ * D_;

  f32_to_bf16_kernel<<<512, 256, 0, stream>>>(Xe, Xeb, LQ_ * D_ / 4);
  f32_to_bf16_kernel<<<512, 256, 0, stream>>>(Hh, Hhb, LK_ * D_ / 4);
  f32_to_bf16_kernel<<<256, 256, 0, stream>>>(Wq, Wqb, D_ * D_ / 4);
  f32_to_bf16_kernel<<<256, 256, 0, stream>>>(Wk, Wkb, D_ * D_ / 4);
  f32_to_bf16_kernel<<<256, 256, 0, stream>>>(Wv, Wvb, D_ * D_ / 4);
  f32_to_bf16_kernel<<<256, 256, 0, stream>>>(Wo, Wob, D_ * D_ / 4);

  const int gemm_blocks = (LQ_ / 32) * (D_ / 64) / 8;  // 128 blocks of 8 waves
  // Q projection with attention scale 1/sqrt(64) folded in (exact in bf16).
  gemm_bf16_wmma<<<gemm_blocks, 256, 0, stream>>>(Xeb, Wqb, Qb, nullptr,
                                                  LQ_, D_, D_, 0.125f, 0);
  gemm_bf16_wmma<<<gemm_blocks, 256, 0, stream>>>(Hhb, Wkb, Kbb, nullptr,
                                                  LK_, D_, D_, 1.0f, 0);
  gemm_bf16_wmma<<<gemm_blocks, 256, 0, stream>>>(Hhb, Wvb, Vtb, nullptr,
                                                  LK_, D_, D_, 1.0f, 1);

  flash_attn_wmma<<<(H_ * (LQ_ / 16)) / 8, 256, 0, stream>>>(Qb, Kbb, Vtb, Ctx);

  gemm_bf16_wmma<<<gemm_blocks, 256, 0, stream>>>(Ctx, Wob, nullptr, out,
                                                  LQ_, D_, D_, 1.0f, 2);
}
